// GraphAttention_34591666602582
// MI455X (gfx1250) — compile-verified, matrix operand order checked by reading, not run on hardware
//
#include <hip/hip_runtime.h>
#include <hip/hip_bf16.h>

// GAT forward, MI455X (gfx1250, wave32, WMMA).
// Roofline: adj (134 MB int32) dominates HBM traffic (~5.8 us @ 23.3 TB/s);
// FLOPs (9.2 GF) trivial vs WMMA ceiling -> fuse softmax + attn@hidden so the
// [8,2048,2048] attention matrix never hits HBM. All hot-loop VMEM is b128.

#define LRELU_ALPHA 0.2f
#define NEG_BIG    -1e12f

typedef __attribute__((ext_vector_type(2)))  float  v2f;
typedef __attribute__((ext_vector_type(8)))  float  v8f;
typedef __attribute__((ext_vector_type(8)))  __bf16 v8bf;
typedef __attribute__((ext_vector_type(16))) __bf16 v16bf;

constexpr int B_ = 8;
constexpr int N_ = 2048;
constexpr int C_ = 128;

// ---------------------------------------------------------------------------
// Kernel 1: hidden = x @ W  via V_WMMA_F32_16X16X4_F32 (fp32-exact path).
// One wave = 16x128 tile, K-loop of 32. Outputs:
//   hidden (fp32, row-major)  -> scores kernel
//   hT     (bf16, [B][C][N] transposed) -> contiguous-K B-frags in kernel 3
// ---------------------------------------------------------------------------
__global__ void gat_hidden_kernel(const float* __restrict__ x,
                                  const float* __restrict__ W,
                                  float* __restrict__ hidden,
                                  __bf16* __restrict__ hT)
{
  const int wave = threadIdx.x >> 5;
  const int lane = threadIdx.x & 31;
  const int tile = blockIdx.x * (blockDim.x >> 5) + wave;   // over (B*N)/16
  const int row0 = tile * 16;
  const int lm   = lane & 15;
  const int koff = (lane >> 4) * 2;

  v8f acc[8] = {};
  for (int k0 = 0; k0 < C_; k0 += 4) {
    v2f a;
    const float* xr = x + (size_t)(row0 + lm) * C_ + (k0 + koff);
    a[0] = xr[0];
    a[1] = xr[1];
#pragma unroll
    for (int nt = 0; nt < 8; ++nt) {
      v2f bv;
      bv[0] = W[(size_t)(k0 + koff    ) * C_ + nt * 16 + lm];
      bv[1] = W[(size_t)(k0 + koff + 1) * C_ + nt * 16 + lm];
      acc[nt] = __builtin_amdgcn_wmma_f32_16x16x4_f32(
          false, a, false, bv, (short)0, acc[nt], false, false);
    }
  }

  const int hl = lane >> 4;
  const int bG = row0 / N_;          // batch of this tile (tiles never straddle)
  const int n0 = row0 % N_;
#pragma unroll
  for (int nt = 0; nt < 8; ++nt)
#pragma unroll
    for (int r = 0; r < 8; ++r) {
      const int nr  = r + 8 * hl;            // row within tile
      const int col = nt * 16 + lm;
      const float v = acc[nt][r];
      hidden[(size_t)(row0 + nr) * C_ + col] = v;
      hT[((size_t)bG * C_ + col) * N_ + (n0 + nr)] = (__bf16)v;
    }
}

// ---------------------------------------------------------------------------
// Kernel 2: scores = hidden @ a  -> attn1[B*N], attn2[B*N]  (tiny)
// ---------------------------------------------------------------------------
__global__ void gat_scores_kernel(const float* __restrict__ hidden,
                                  const float* __restrict__ a,
                                  float* __restrict__ attn1,
                                  float* __restrict__ attn2)
{
  __shared__ float s1[128];
  __shared__ float s2[128];
  const int r = blockIdx.x;
  const int t = threadIdx.x;
  const float h = hidden[(size_t)r * C_ + t];
  s1[t] = h * a[t * 2 + 0];
  s2[t] = h * a[t * 2 + 1];
  __syncthreads();
  for (int off = 64; off > 0; off >>= 1) {
    if (t < off) { s1[t] += s1[t + off]; s2[t] += s2[t + off]; }
    __syncthreads();
  }
  if (t == 0) { attn1[r] = s1[0]; attn2[r] = s2[0]; }
}

// ---------------------------------------------------------------------------
// Kernel 3: fused mask + leaky-relu + softmax + attn@hidden + bias.
// One wave per 16-row tile.
//  Pass 1: per-row max; lane owns 4 consecutive columns (int4/float4 loads),
//          16 running maxima in registers, shfl-xor reduce, prefetch ahead.
//  Pass 2: un-normalized P=exp(e-m) built straight into the 16-bit A-frag
//          layout; B-frags are two b128 loads from transposed hT; 8x
//          v_wmma_f32_16x16x32_bf16 per 32-col chunk; rescale+bias at store.
// ---------------------------------------------------------------------------
__global__ void gat_attn_kernel(const int* __restrict__ adj,
                                const float* __restrict__ attn1,
                                const float* __restrict__ attn2,
                                const __bf16* __restrict__ hT,
                                const float* __restrict__ bias,
                                float* __restrict__ out)
{
  __shared__ float sM[4][16];   // per-wave row maxima
  __shared__ float sL[4][16];   // per-wave 1/rowsum

  const int wave = threadIdx.x >> 5;
  const int lane = threadIdx.x & 31;
  const int tile = blockIdx.x * (blockDim.x >> 5) + wave;   // over B*(N/16)
  const int bI   = tile / (N_ / 16);
  const int i0   = (tile % (N_ / 16)) * 16;

  const int*   adjr = adj   + ((size_t)bI * N_ + i0) * N_;  // 16 x N block
  const float* a2   = attn2 + (size_t)bI * N_;

  // ---- pass 1: per-row max of masked leaky-relu logits ----
  float a1r[16], mloc[16];
#pragma unroll
  for (int mm = 0; mm < 16; ++mm) {
    a1r[mm]  = attn1[(size_t)bI * N_ + i0 + mm];
    mloc[mm] = -3.0e38f;
  }
  for (int jb = 0; jb < N_; jb += 128) {
    const int j0 = jb + lane * 4;
    const float4 a2v = *(const float4*)(a2 + j0);
#pragma unroll
    for (int mm = 0; mm < 16; ++mm) {
      if (mm == 0 && jb + 128 < N_)
        __builtin_prefetch(adjr + j0 + 128, 0, 1);   // global_prefetch_b8
      const int4 av = *(const int4*)(adjr + (size_t)mm * N_ + j0);
      float e0 = a1r[mm] + a2v.x; e0 = (e0 >= 0.f) ? e0 : LRELU_ALPHA * e0;
      float e1 = a1r[mm] + a2v.y; e1 = (e1 >= 0.f) ? e1 : LRELU_ALPHA * e1;
      float e2 = a1r[mm] + a2v.z; e2 = (e2 >= 0.f) ? e2 : LRELU_ALPHA * e2;
      float e3 = a1r[mm] + a2v.w; e3 = (e3 >= 0.f) ? e3 : LRELU_ALPHA * e3;
      e0 = (av.x > 0) ? e0 : e0 + NEG_BIG;
      e1 = (av.y > 0) ? e1 : e1 + NEG_BIG;
      e2 = (av.z > 0) ? e2 : e2 + NEG_BIG;
      e3 = (av.w > 0) ? e3 : e3 + NEG_BIG;
      mloc[mm] = fmaxf(mloc[mm], fmaxf(fmaxf(e0, e1), fmaxf(e2, e3)));
    }
  }
#pragma unroll
  for (int mm = 0; mm < 16; ++mm) {
    float mv = mloc[mm];
#pragma unroll
    for (int s = 16; s > 0; s >>= 1)
      mv = fmaxf(mv, __shfl_xor(mv, s, 32));
    if (lane == mm) sM[wave][mm] = mv;
  }
  __syncthreads();

  // ---- pass 2: P-tile build + WMMA accumulate (adj re-read hits L2) ----
  const int lm = lane & 15;            // row M for A-frag / col N for B-frag
  const int kb = (lane >> 4) * 8;      // K base (hi half-wave carries K+8)
  const float a1m = attn1[(size_t)bI * N_ + i0 + lm];
  const float rM  = sM[wave][lm];
  const int*   adjm = adjr + (size_t)lm * N_;
  const __bf16* hTb = hT + (size_t)bI * C_ * N_;

  float ssum = 0.f;
  v8f acc[8] = {};

  for (int jc = 0; jc < N_; jc += 32) {
    // A-frag: lane's K set is {kb..kb+7} U {16+kb..16+kb+7} -> 4x b128 loads
    const int4   ad0 = *(const int4*)  (adjm + jc + kb);
    const int4   ad1 = *(const int4*)  (adjm + jc + kb + 4);
    const int4   ad2 = *(const int4*)  (adjm + jc + 16 + kb);
    const int4   ad3 = *(const int4*)  (adjm + jc + 16 + kb + 4);
    const float4 s0  = *(const float4*)(a2 + jc + kb);
    const float4 s1  = *(const float4*)(a2 + jc + kb + 4);
    const float4 s2  = *(const float4*)(a2 + jc + 16 + kb);
    const float4 s3  = *(const float4*)(a2 + jc + 16 + kb + 4);

    v16bf A;
#define P_ELT(idx, a2v, adv)                                        \
    { float e = a1m + (a2v);                                        \
      e = (e >= 0.f) ? e : LRELU_ALPHA * e;                         \
      e = ((adv) > 0) ? e : e + NEG_BIG;                            \
      const float p = __expf(e - rM);                               \
      ssum += p;                                                    \
      A[idx] = (__bf16)p; }
    P_ELT(0,  s0.x, ad0.x)  P_ELT(1,  s0.y, ad0.y)
    P_ELT(2,  s0.z, ad0.z)  P_ELT(3,  s0.w, ad0.w)
    P_ELT(4,  s1.x, ad1.x)  P_ELT(5,  s1.y, ad1.y)
    P_ELT(6,  s1.z, ad1.z)  P_ELT(7,  s1.w, ad1.w)
    P_ELT(8,  s2.x, ad2.x)  P_ELT(9,  s2.y, ad2.y)
    P_ELT(10, s2.z, ad2.z)  P_ELT(11, s2.w, ad2.w)
    P_ELT(12, s3.x, ad3.x)  P_ELT(13, s3.y, ad3.y)
    P_ELT(14, s3.z, ad3.z)  P_ELT(15, s3.w, ad3.w)
#undef P_ELT

#pragma unroll
    for (int nt = 0; nt < 8; ++nt) {
      // B-frag: hT[(nt*16+lm)][jc + K] -> K-contiguous => two b128 loads
      const __bf16* hrow = hTb + (size_t)(nt * 16 + lm) * N_ + jc;
      const v8bf lo = *(const v8bf*)(hrow + kb);
      const v8bf hi = *(const v8bf*)(hrow + 16 + kb);
      const v16bf Bv = __builtin_shufflevector(lo, hi,
          0, 1, 2, 3, 4, 5, 6, 7, 8, 9, 10, 11, 12, 13, 14, 15);
      acc[nt] = __builtin_amdgcn_wmma_f32_16x16x32_bf16(
          false, A, false, Bv, (short)0, acc[nt], false, false);
    }
  }

  // row sums: row m lives in lanes m and m+16 (they split the K range)
  ssum += __shfl_xor(ssum, 16, 32);
  if (lane < 16) sL[wave][lane] = 1.f / ssum;
  __syncthreads();

  // ---- store: normalize + bias; C/D layout: VGPR r -> M=r (+8 hi lanes) ----
  const int hl = lane >> 4;
#pragma unroll
  for (int nt = 0; nt < 8; ++nt)
#pragma unroll
    for (int r = 0; r < 8; ++r) {
      const int row = i0 + r + 8 * hl;
      const int col = nt * 16 + lm;
      const float scale = sL[wave][r + 8 * hl];
      out[((size_t)bI * N_ + row) * C_ + col] = acc[nt][r] * scale + bias[col];
    }
}

// ---------------------------------------------------------------------------
extern "C" void kernel_launch(void* const* d_in, const int* in_sizes, int n_in,
                              void* d_out, int out_size, void* d_ws, size_t ws_size,
                              hipStream_t stream) {
  (void)in_sizes; (void)n_in; (void)out_size; (void)ws_size;
  const float* x    = (const float*)d_in[0];
  const int*   adj  = (const int*)  d_in[1];
  const float* W    = (const float*)d_in[2];
  const float* a    = (const float*)d_in[3];
  const float* bias = (const float*)d_in[4];
  float* out = (float*)d_out;

  char* ws = (char*)d_ws;
  float*  hidden = (float*) ws;                                   // 8 MB fp32
  __bf16* hT     = (__bf16*)(ws + (size_t)8  * 1024 * 1024);      // 4 MB bf16 (transposed)
  float*  attn1  = (float*) (ws + (size_t)12 * 1024 * 1024);      // 64 KB
  float*  attn2  = attn1 + (size_t)B_ * N_;                       // 64 KB

  // 1024 row-tiles, 8 waves/block
  gat_hidden_kernel<<<(B_ * N_ / 16) / 8, 256, 0, stream>>>(x, W, hidden, hT);
  // one block per row
  gat_scores_kernel<<<B_ * N_, 128, 0, stream>>>(hidden, a, attn1, attn2);
  // 1024 row-tiles, 4 waves/block
  gat_attn_kernel<<<(B_ * (N_ / 16)) / 4, 128, 0, stream>>>(adj, attn1, attn2,
                                                            hT, bias, out);
}
